// MultiHeadAttentionQuantum_65481071399608
// MI455X (gfx1250) — compile-verified, matrix operand order checked by reading, not run on hardware
//
#include <hip/hip_runtime.h>

// ---------------------------------------------------------------------------
// Types & helpers
// ---------------------------------------------------------------------------
typedef __attribute__((ext_vector_type(4)))  _Float16 v4h;
typedef __attribute__((ext_vector_type(8)))  _Float16 v8h;
typedef __attribute__((ext_vector_type(16))) _Float16 v16h;
typedef __attribute__((ext_vector_type(8)))  float    v8f;

union F16Frag { v16h v; v8h h[2]; };

#define WMMA_F16(A, B, C) \
  __builtin_amdgcn_wmma_f32_16x16x32_f16(false, (A), false, (B), (short)0, (C), false, false)

// Load a 16x32 f16 fragment (A-style or B-style; both use the same per-lane
// pattern when the source is row-major [16 rows, K contiguous]):
//   lane l<16  : row = row0+l,      halves K = {k0+0..7,  k0+16..23}
//   lane l>=16 : row = row0+(l-16), halves K = {k0+8..15, k0+24..31}
__device__ __forceinline__ v16h load_frag_rm(const _Float16* base, int ld,
                                             int row0, int col0, int lane) {
  const int lr = lane & 15;
  const int hf = lane >> 4;
  const _Float16* p = base + (size_t)(row0 + lr) * (size_t)ld + col0 + hf * 8;
  F16Frag f;
  f.h[0] = *(const v8h*)(p);
  f.h[1] = *(const v8h*)(p + 16);
  return f.v;
}

// Problem constants
#define S_LEN   2048
#define BATCH   2
#define EMB     1024
#define HEADS   16
#define HD      64
#define ROWS    (BATCH * S_LEN)   // 4096
#define N_QKV   (3 * EMB)         // 3072

#define INV_2PI 0.15915494309189535f
#define LOG2E   1.4426950408889634f

// ---------------------------------------------------------------------------
// f32 -> f16 conversion (4 elements / thread)
// ---------------------------------------------------------------------------
__global__ void __launch_bounds__(256) cvt_f32_to_f16(const float* __restrict__ src,
                                                      _Float16* __restrict__ dst, int n) {
  int idx = (blockIdx.x * blockDim.x + threadIdx.x) * 4;
  if (idx < n) {
    v4h o;
    o[0] = (_Float16)src[idx + 0];
    o[1] = (_Float16)src[idx + 1];
    o[2] = (_Float16)src[idx + 2];
    o[3] = (_Float16)src[idx + 3];
    *(v4h*)(dst + idx) = o;
  }
}

// ---------------------------------------------------------------------------
// Tiled WMMA GEMM:  C[M,N] = A[M,K] * B[N,K]^T  (both f16 row-major, K contig)
// Wave tile 64x32 (4x2 tiles of 16x16), block = 4 waves covering N=128.
// EPI==0: plain f32 store to outf.
// EPI==1: quantum activation cos(v)*cos(qparams[n&63]); split into q / k / vT.
// ---------------------------------------------------------------------------
template <int EPI>
__global__ void __launch_bounds__(128)
gemm_wmma(const _Float16* __restrict__ A, const _Float16* __restrict__ Bm,
          int M, int N, int K,
          const float* __restrict__ qparams,
          _Float16* __restrict__ qdst, _Float16* __restrict__ kdst,
          _Float16* __restrict__ vTdst, float* __restrict__ outf) {
  const int lane = threadIdx.x & 31;
  const int wave = threadIdx.x >> 5;
  const int n0 = blockIdx.x * 128 + wave * 32;   // wave's 32 output columns
  const int m0 = blockIdx.y * 64;                // wave's 64 output rows

  v8f acc[4][2];
#pragma unroll
  for (int i = 0; i < 4; ++i)
#pragma unroll
    for (int j = 0; j < 2; ++j) acc[i][j] = {};

  for (int k0 = 0; k0 < K; k0 += 32) {
    // prefetch next K tile into cache (global_prefetch_b8)
    if (k0 + 32 < K) {
      __builtin_prefetch(A + (size_t)(m0 + (lane & 15)) * K + k0 + 32, 0, 3);
      __builtin_prefetch(Bm + (size_t)(n0 + (lane & 15)) * K + k0 + 32, 0, 3);
    }
    v16h a0 = load_frag_rm(A, K, m0 +  0, k0, lane);
    v16h a1 = load_frag_rm(A, K, m0 + 16, k0, lane);
    v16h a2 = load_frag_rm(A, K, m0 + 32, k0, lane);
    v16h a3 = load_frag_rm(A, K, m0 + 48, k0, lane);
    v16h b0 = load_frag_rm(Bm, K, n0 +  0, k0, lane);
    v16h b1 = load_frag_rm(Bm, K, n0 + 16, k0, lane);

    acc[0][0] = WMMA_F16(a0, b0, acc[0][0]);
    acc[0][1] = WMMA_F16(a0, b1, acc[0][1]);
    acc[1][0] = WMMA_F16(a1, b0, acc[1][0]);
    acc[1][1] = WMMA_F16(a1, b1, acc[1][1]);
    acc[2][0] = WMMA_F16(a2, b0, acc[2][0]);
    acc[2][1] = WMMA_F16(a2, b1, acc[2][1]);
    acc[3][0] = WMMA_F16(a3, b0, acc[3][0]);
    acc[3][1] = WMMA_F16(a3, b1, acc[3][1]);
  }

  // Epilogue. C-tile layout: lane holds column n = tile_n + (lane&15),
  // VGPR r holds row m = tile_m + r + (lane>=16 ? 8 : 0).
  const int lr = lane & 15;
  const int hf = lane >> 4;
#pragma unroll
  for (int i = 0; i < 4; ++i) {
#pragma unroll
    for (int j = 0; j < 2; ++j) {
      const int n  = n0 + j * 16 + lr;
      const int mb = m0 + i * 16 + hf * 8;
      if (EPI == 0) {
#pragma unroll
        for (int r = 0; r < 8; ++r)
          outf[(size_t)(mb + r) * N + n] = acc[i][j][r];
      } else {
        const float cq = __builtin_amdgcn_cosf(qparams[n & 63] * INV_2PI);
        if (n < 2 * EMB) {
          // q or k: activated f16 at [row, n % 1024]
          _Float16* d = (n < EMB) ? (qdst + n) : (kdst + (n - EMB));
#pragma unroll
          for (int r = 0; r < 8; ++r) {
            float v = __builtin_amdgcn_cosf(acc[i][j][r] * INV_2PI) * cq;
            d[(size_t)(mb + r) * EMB] = (_Float16)v;
          }
        } else {
          // v: store TRANSPOSED as vT[b*1024 + (h*64+d)][s] for fast PV B-frags
          const int vr = n - 2 * EMB;        // h*64 + d  in [0,1024)
          const int b  = mb >> 11;           // mb / 2048
          const int s  = mb & 2047;
          v8h pk;
#pragma unroll
          for (int r = 0; r < 8; ++r)
            pk[r] = (_Float16)(__builtin_amdgcn_cosf(acc[i][j][r] * INV_2PI) * cq);
          *(v8h*)(vTdst + ((size_t)(b * EMB + vr)) * S_LEN + s) = pk;
        }
      }
    }
  }
}

// ---------------------------------------------------------------------------
// Flash attention with STATIC-MAX softmax.
//   q,k entries are cos(.)*cos(.) in [-1,1]  =>  |score|/sqrt(64) <= 8.
// So exp2(s*log2e/8 - 8*log2e) is in [2^-23.1, 1]: no overflow possible, the
// static bias cancels in the softmax ratio, and NO online max / rescale is
// needed. Row sums accumulate as per-lane partials; one cross-lane reduction
// runs at the end instead of per key-block.
// One wave = 16 query rows of one (b,h). P converts C-layout -> A-fragment
// layout through a 1KB/wave private LDS region. 8 WMMAs per 32-key block.
// ---------------------------------------------------------------------------
__global__ void __launch_bounds__(256)
attn_flash(const _Float16* __restrict__ qbuf, const _Float16* __restrict__ kbuf,
           const _Float16* __restrict__ vT, _Float16* __restrict__ abuf) {
  __shared__ _Float16 lds_p[8][16 * 32];   // 8 waves * 1KB

  const int lane = threadIdx.x & 31;
  const int wave = threadIdx.x >> 5;
  const int bh = blockIdx.x;               // 0..31
  const int b = bh >> 4, h = bh & 15;
  const int q0 = blockIdx.y * 128 + wave * 16;  // query row within S
  const int rowbase = b * S_LEN;
  const int lr = lane & 15;
  const int hf = lane >> 4;

  // Q fragments for d-slices [0,32) and [32,64)
  const v16h qf0 = load_frag_rm(qbuf, EMB, rowbase + q0, h * HD + 0,  lane);
  const v16h qf1 = load_frag_rm(qbuf, EMB, rowbase + q0, h * HD + 32, lane);

  v8f o[4];
#pragma unroll
  for (int dt = 0; dt < 4; ++dt) o[dt] = {};
  float psum[8];
#pragma unroll
  for (int r = 0; r < 8; ++r) psum[r] = 0.0f;

  const float SC   = 0.125f * LOG2E;  // 1/sqrt(64) folded into log2 domain
  const float SMAX = 8.0f * LOG2E;    // static score bound (|s|<=8 by construction)
  _Float16* myp = &lds_p[wave][0];
  const _Float16* vbase = vT + ((size_t)(b * EMB + h * HD)) * S_LEN;

  for (int kb = 0; kb < S_LEN; kb += 32) {
    // scores for 32 keys = two 16x16 tiles, K-dim = D = 64 (2 WMMA steps each)
    v16h k00 = load_frag_rm(kbuf, EMB, rowbase + kb +  0, h * HD + 0,  lane);
    v16h k01 = load_frag_rm(kbuf, EMB, rowbase + kb +  0, h * HD + 32, lane);
    v16h k10 = load_frag_rm(kbuf, EMB, rowbase + kb + 16, h * HD + 0,  lane);
    v16h k11 = load_frag_rm(kbuf, EMB, rowbase + kb + 16, h * HD + 32, lane);

    v8f s0 = {}; v8f s1 = {};
    s0 = WMMA_F16(qf0, k00, s0);
    s0 = WMMA_F16(qf1, k01, s0);
    s1 = WMMA_F16(qf0, k10, s1);
    s1 = WMMA_F16(qf1, k11, s1);

    // static-max softmax numerator + per-lane partial row sums
    float p0[8], p1[8];
#pragma unroll
    for (int r = 0; r < 8; ++r) {
      p0[r] = __builtin_amdgcn_exp2f(s0[r] * SC - SMAX);
      p1[r] = __builtin_amdgcn_exp2f(s1[r] * SC - SMAX);
      psum[r] += p0[r] + p1[r];
    }

    // C-layout P -> row-major 16x32 f16 in this wave's private LDS region
#pragma unroll
    for (int r = 0; r < 8; ++r) {
      const int mrow = r + hf * 8;
      myp[mrow * 32 + lr]      = (_Float16)p0[r];
      myp[mrow * 32 + 16 + lr] = (_Float16)p1[r];
    }
    // reload as an A-fragment (same-wave DS ops are in order)
    v16h pf;
    {
      const _Float16* pp = myp + lr * 32 + hf * 8;
      F16Frag f;
      f.h[0] = *(const v8h*)(pp);
      f.h[1] = *(const v8h*)(pp + 16);
      pf = f.v;
    }

    // O += P(16x32) * V(32 x 64); vT rows are [d][s] so B-frags are contiguous
#pragma unroll
    for (int dt = 0; dt < 4; ++dt) {
      v16h vf = load_frag_rm(vbase, S_LEN, dt * 16, kb, lane);
      o[dt] = WMMA_F16(pf, vf, o[dt]);
    }
  }

  // One cross-lane row-sum reduction at the end (masks 1/2/4/8 stay inside
  // the 16-lane half that owns each row in the C-tile layout).
  float rinv[8];
#pragma unroll
  for (int r = 0; r < 8; ++r) {
    float ps = psum[r];
#pragma unroll
    for (int m = 1; m < 16; m <<= 1) ps += __shfl_xor(ps, m, 32);
    rinv[r] = __builtin_amdgcn_rcpf(ps);
  }

  // normalize + store f16 attention output at [b*S+s, h*64+d]
#pragma unroll
  for (int dt = 0; dt < 4; ++dt) {
    const int n = h * HD + dt * 16 + lr;
#pragma unroll
    for (int r = 0; r < 8; ++r) {
      const int m = q0 + hf * 8 + r;
      abuf[(size_t)(rowbase + m) * EMB + n] = (_Float16)(o[dt][r] * rinv[r]);
    }
  }
}

// ---------------------------------------------------------------------------
// Launch
// ---------------------------------------------------------------------------
extern "C" void kernel_launch(void* const* d_in, const int* in_sizes, int n_in,
                              void* d_out, int out_size, void* d_ws, size_t ws_size,
                              hipStream_t stream) {
  const float* x       = (const float*)d_in[0];   // [2,2048,1024]
  const float* qkv_w   = (const float*)d_in[1];   // [3072,1024]
  const float* out_w   = (const float*)d_in[2];   // [1024,1024]
  const float* qparams = (const float*)d_in[3];   // [64]
  float* out = (float*)d_out;                     // [2,2048,1024] f32

  char* ws = (char*)d_ws;
  const size_t MB = 1u << 20;
  _Float16* xh    = (_Float16*)(ws +  0 * MB);  //  8 MiB  x as f16
  _Float16* wqkvh = (_Float16*)(ws +  8 * MB);  //  6 MiB  qkv_w as f16
  _Float16* wouth = (_Float16*)(ws + 14 * MB);  //  2 MiB  out_w as f16
  _Float16* qbuf  = (_Float16*)(ws + 16 * MB);  //  8 MiB  activated q [4096,1024]
  _Float16* kbuf  = (_Float16*)(ws + 24 * MB);  //  8 MiB  activated k [4096,1024]
  _Float16* vTbuf = (_Float16*)(ws + 32 * MB);  //  8 MiB  activated v^T [2*1024,2048]
  _Float16* abuf  = (_Float16*)(ws + 40 * MB);  //  8 MiB  attention out [4096,1024]

  // 1) precision conversions
  cvt_f32_to_f16<<<(ROWS * EMB) / 1024, 256, 0, stream>>>(x, xh, ROWS * EMB);
  cvt_f32_to_f16<<<(N_QKV * EMB) / 1024, 256, 0, stream>>>(qkv_w, wqkvh, N_QKV * EMB);
  cvt_f32_to_f16<<<(EMB * EMB) / 1024, 256, 0, stream>>>(out_w, wouth, EMB * EMB);

  // 2) fused QKV GEMM + quantum activation (q,k normal; v transposed)
  {
    dim3 grid(N_QKV / 128, ROWS / 64);
    gemm_wmma<1><<<grid, 128, 0, stream>>>(xh, wqkvh, ROWS, N_QKV, EMB,
                                           qparams, qbuf, kbuf, vTbuf, nullptr);
  }

  // 3) flash attention: 32 (b,h) pairs x 16 chunks of 128 query rows
  {
    dim3 grid(BATCH * HEADS, S_LEN / 128);
    attn_flash<<<grid, 256, 0, stream>>>(qbuf, kbuf, vTbuf, abuf);
  }

  // 4) output projection GEMM, f32 epilogue to d_out
  {
    dim3 grid(EMB / 128, ROWS / 64);
    gemm_wmma<0><<<grid, 128, 0, stream>>>(abuf, wouth, ROWS, EMB, EMB,
                                           nullptr, nullptr, nullptr, nullptr, out);
  }
}
